// GCN2_59296318488677
// MI455X (gfx1250) — matrix operand order repaired and checked by reading, not compiled, not run on hardware
//
#include <hip/hip_runtime.h>
#include <hip/hip_bf16.h>
#include <math.h>

typedef __attribute__((ext_vector_type(16))) __bf16 v16bf;
typedef __attribute__((ext_vector_type(8)))  float  v8f;

#define TK   32
#define LDST (TK + 8)   // bf16 row stride: 40 elems = 80 B (multiple of 16 B)

// ---------------------------------------------------------------------------
// WMMA GEMM:  Y[M,Nout] = X[M,K] @ W[Nout,K]^T  (+bias)(+add)(ReLU)
// Block = 256 threads = 8 wave32. Block tile = 128 rows x Nout cols.
// f32 -> bf16 conversion happens once, during the global->LDS stage; WMMA
// fragments are then loaded as contiguous 16-byte LDS chunks (ds_load_b128).
// ---------------------------------------------------------------------------
template <int NTILES>
__global__ __launch_bounds__(256)
void gemm_wmma(const float* __restrict__ X, const float* __restrict__ W,
               const float* __restrict__ bias, const float* __restrict__ add,
               float* __restrict__ Y, int M, int K, int Nout, int relu)
{
  __shared__ __bf16 Asb[128 * LDST];
  __shared__ __bf16 Bsb[128 * LDST];
  const int tid  = threadIdx.x;
  const int wave = tid >> 5;
  const int lane = tid & 31;
  const int hl   = lane >> 4;   // 16-lane half
  const int l16  = lane & 15;
  const int mbase = blockIdx.x * 128;

  v8f acc[NTILES] = {};

  for (int k0 = 0; k0 < K; k0 += TK) {
    // stage A tile (128 x 32) as bf16; float2 global loads, packed converts
    for (int i = tid; i < 128 * (TK / 2); i += 256) {
      const int r = i >> 4, c2 = (i & 15) << 1;
      const float2 v = *(const float2*)(X + (long)(mbase + r) * K + k0 + c2);
      __bf16* dst = &Asb[r * LDST + c2];
      dst[0] = (__bf16)v.x; dst[1] = (__bf16)v.y;
    }
    // stage B tile (Nout x 32) as bf16
    for (int i = tid; i < Nout * (TK / 2); i += 256) {
      const int r = i >> 4, c2 = (i & 15) << 1;
      const float2 v = *(const float2*)(W + (long)r * K + k0 + c2);
      __bf16* dst = &Bsb[r * LDST + c2];
      dst[0] = (__bf16)v.x; dst[1] = (__bf16)v.y;
    }
    __syncthreads();

    // A fragment (16x32): afrag[e] = A[row][8*hl + e], A[row][16 + 8*hl + e]
    v16bf afrag;
    {
      const __bf16* a0 = &Asb[(wave * 16 + l16) * LDST + 8 * hl];
      const __bf16* a1 = a0 + 16;
      #pragma unroll
      for (int e = 0; e < 8; ++e) { afrag[e] = a0[e]; afrag[8 + e] = a1[e]; }
    }

    #pragma unroll
    for (int t = 0; t < NTILES; ++t) {
      // B fragment (32x16): bfrag[e] = W[nb + lane%16][8*hl + e], W[nb + lane%16][16 + 8*hl + e]
      const __bf16* bp = &Bsb[(t * 16 + l16) * LDST + 8 * hl];
      v16bf bfrag;
      #pragma unroll
      for (int e = 0; e < 8; ++e) { bfrag[e] = bp[e]; bfrag[8 + e] = bp[16 + e]; }
      acc[t] = __builtin_amdgcn_wmma_f32_16x16x32_bf16(
          false, afrag, false, bfrag, (short)0, acc[t], false, false);
    }
    __syncthreads();
  }

  // Epilogue. D layout: VGPR r -> M = wave*16 + r + 8*half, N = nb + lane%16.
  #pragma unroll
  for (int t = 0; t < NTILES; ++t) {
    const int n = t * 16 + l16;
    #pragma unroll
    for (int r = 0; r < 8; ++r) {
      const int m = mbase + wave * 16 + r + hl * 8;
      float v = acc[t][r];
      if (bias) v += bias[n];
      if (add)  v += add[(long)m * Nout + n];
      if (relu) v = fmaxf(v, 0.0f);
      Y[(long)m * Nout + n] = v;
    }
  }
}

// ---------------------------------------------------------------------------
// Utility / elementwise kernels
// ---------------------------------------------------------------------------
__global__ void fill_f32(float* a, float v, int n) {
  int i = blockIdx.x * blockDim.x + threadIdx.x;
  if (i < n) a[i] = v;
}
__global__ void fill_i32(int* a, int v, int n) {
  int i = blockIdx.x * blockDim.x + threadIdx.x;
  if (i < n) a[i] = v;
}
__global__ void copy_i32(int* dst, const int* src, int n) {
  int i = blockIdx.x * blockDim.x + threadIdx.x;
  if (i < n) dst[i] = src[i];
}

// edge scatter: agg[dst] += x[src] * emask   (4 features per thread)
__global__ void scatter_edges(const float* __restrict__ x, const int* __restrict__ ei,
                              const float* __restrict__ emask, float* __restrict__ agg,
                              int E_, int din)
{
  const int chunks = din >> 2;
  int idx = blockIdx.x * blockDim.x + threadIdx.x;
  if (idx >= E_ * chunks) return;
  int e = idx / chunks;
  int c = (idx - e * chunks) << 2;
  float m = emask[e];
  if (m == 0.0f) return;
  int s = ei[e], d = ei[E_ + e];
  const float* xs = x + (long)s * din + c;
  __builtin_prefetch(xs, 0, 0);              // global_prefetch_b8
  float* ag = agg + (long)d * din + c;
  atomicAdd(ag + 0, xs[0] * m);
  atomicAdd(ag + 1, xs[1] * m);
  atomicAdd(ag + 2, xs[2] * m);
  atomicAdd(ag + 3, xs[3] * m);
}

// batch-norm stats: one block per column (128 columns)
__global__ void bn_stats(const float* __restrict__ x, float* mean, float* var, int M)
{
  const int c = blockIdx.x;
  float s = 0.f, s2 = 0.f;
  for (int r = threadIdx.x; r < M; r += 256) {
    float v = x[(long)r * 128 + c];
    s += v; s2 += v * v;
  }
  __shared__ float sh[256], sh2[256];
  sh[threadIdx.x] = s; sh2[threadIdx.x] = s2;
  __syncthreads();
  for (int o = 128; o > 0; o >>= 1) {
    if (threadIdx.x < o) { sh[threadIdx.x] += sh[threadIdx.x + o];
                           sh2[threadIdx.x] += sh2[threadIdx.x + o]; }
    __syncthreads();
  }
  if (threadIdx.x == 0) {
    float m = sh[0] / (float)M;
    mean[c] = m;
    var[c] = sh2[0] / (float)M - m * m;
  }
}

__global__ void bn_apply(float* x, const float* mean, const float* var,
                         const float* g, const float* be, int M)
{
  long i = (long)blockIdx.x * 256 + threadIdx.x;
  if (i >= (long)M * 128) return;
  int c = (int)(i & 127);
  x[i] = (x[i] - mean[c]) * __frsqrt_rn(var[c] + 1e-5f) * g[c] + be[c];
}

__global__ void pnorm_k(const float* p, float* pn)
{
  if (threadIdx.x == 0 && blockIdx.x == 0) {
    float s = 0.f;
    for (int c = 0; c < 128; ++c) s += p[c] * p[c];
    pn[0] = sqrtf(s);
  }
}

__global__ void score_k(const float* __restrict__ x, const float* __restrict__ p,
                        const float* __restrict__ pn, float* score, int M)
{
  int i = blockIdx.x * blockDim.x + threadIdx.x;
  if (i >= M) return;
  const float* xr = x + (long)i * 128;
  float s = 0.f;
  for (int c = 0; c < 128; ++c) s += xr[c] * p[c];
  score[i] = s / pn[0];
}

// per-graph descending bitonic sort of (score,idx) in LDS; emit top-k indices
__global__ void topk_k(const float* __restrict__ score, int* __restrict__ perm,
                       int ncur, int k)
{
  __shared__ float sv[512];
  __shared__ int   si[512];
  const int g = blockIdx.x;
  for (int t = threadIdx.x; t < ncur; t += 256) { sv[t] = score[g * ncur + t]; si[t] = t; }
  __syncthreads();
  for (int ks = 2; ks <= ncur; ks <<= 1) {
    for (int j = ks >> 1; j > 0; j >>= 1) {
      for (int t = threadIdx.x; t < ncur; t += 256) {
        int xj = t ^ j;
        if (xj > t) {
          bool dir = ((t & ks) == 0);     // descending overall
          float a = sv[t], b = sv[xj];
          bool sw = dir ? (a < b) : (a > b);
          if (sw) {
            sv[t] = b; sv[xj] = a;
            int tmp = si[t]; si[t] = si[xj]; si[xj] = tmp;
          }
        }
      }
      __syncthreads();
    }
  }
  for (int t = threadIdx.x; t < k; t += 256) perm[g * k + t] = g * ncur + si[t];
}

__global__ void gather_gate(const float* __restrict__ x, const float* __restrict__ score,
                            const int* __restrict__ perm, float* __restrict__ out, int Mnew)
{
  long i = (long)blockIdx.x * 256 + threadIdx.x;
  if (i >= (long)Mnew * 128) return;
  int node = (int)(i >> 7), c = (int)(i & 127);
  int o = perm[node];
  out[i] = x[(long)o * 128 + c] * tanhf(score[o]);
}

__global__ void inv_scatter(int* inv, const int* perm, int Mnew)
{
  int i = blockIdx.x * blockDim.x + threadIdx.x;
  if (i < Mnew) inv[perm[i]] = i;
}

__global__ void relabel(int* ei, float* emask, const int* __restrict__ inv, int E_)
{
  int i = blockIdx.x * blockDim.x + threadIdx.x;
  if (i >= E_) return;
  int s = ei[i], d = ei[E_ + i];
  int ns = inv[s], nd = inv[d];
  float m = emask[i];
  if (ns < 0 || nd < 0) m = 0.0f;
  emask[i] = m;
  ei[i]      = ns > 0 ? ns : 0;
  ei[E_ + i] = nd > 0 ? nd : 0;
}

// per-graph max+mean readout, accumulated into rsum[B,256]
__global__ void readout_k(const float* __restrict__ x, float* __restrict__ rsum, int k)
{
  const int g = blockIdx.x;
  const int f = threadIdx.x;   // 128 threads
  float mx = -INFINITY, sm = 0.f;
  for (int j = 0; j < k; ++j) {
    float v = x[((long)(g * k + j)) * 128 + f];
    mx = fmaxf(mx, v);
    sm += v;
  }
  rsum[g * 256 + f]       += mx;
  rsum[g * 256 + 128 + f] += sm / (float)k;
}

// lin3 (10x64) + log_softmax, one block per graph
__global__ void head_k(const float* __restrict__ h2, const float* __restrict__ W3,
                       const float* __restrict__ b3, float* __restrict__ out)
{
  const int g = blockIdx.x;
  const int t = threadIdx.x;   // 32 threads
  __shared__ float lg[10];
  if (t < 10) {
    float s = b3[t];
    for (int c = 0; c < 64; ++c) s += h2[g * 64 + c] * W3[t * 64 + c];
    lg[t] = s;
  }
  __syncthreads();
  if (t == 0) {
    float mx = lg[0];
    for (int i = 1; i < 10; ++i) mx = fmaxf(mx, lg[i]);
    float se = 0.f;
    for (int i = 0; i < 10; ++i) se += expf(lg[i] - mx);
    float lse = mx + logf(se);
    for (int i = 0; i < 10; ++i) out[g * 10 + i] = lg[i] - lse;
  }
}

// ---------------------------------------------------------------------------
// Orchestration
// ---------------------------------------------------------------------------
static inline int gdiv(long n, int b) { return (int)((n + b - 1) / b); }

extern "C" void kernel_launch(void* const* d_in, const int* in_sizes, int n_in,
                              void* d_out, int out_size, void* d_ws, size_t ws_size,
                              hipStream_t stream)
{
  (void)in_sizes; (void)n_in; (void)out_size; (void)ws_size;
  const int N0 = 128 * 512, Ecnt = N0 * 16, B = 128;

  const float* x_in = (const float*)d_in[0];
  const float *Wroot[5], *Wrel[5], *cb[5], *bng[5], *bnb[5], *pp[5];
  for (int i = 0; i < 5; ++i) {
    Wroot[i] = (const float*)d_in[1 + 3 * i];
    Wrel[i]  = (const float*)d_in[2 + 3 * i];
    cb[i]    = (const float*)d_in[3 + 3 * i];
    bng[i]   = (const float*)d_in[16 + 2 * i];
    bnb[i]   = (const float*)d_in[17 + 2 * i];
    pp[i]    = (const float*)d_in[26 + i];
  }
  const float* lin1W = (const float*)d_in[31];
  const float* lin1b = (const float*)d_in[32];
  const float* lin2W = (const float*)d_in[33];
  const float* lin2b = (const float*)d_in[34];
  const float* lin3W = (const float*)d_in[35];
  const float* lin3b = (const float*)d_in[36];
  const int*   eidx  = (const int*)d_in[37];

  // workspace carve
  char* w = (char*)d_ws;
  float* xA    = (float*)w; w += (long)N0 * 128 * 4;        // 32 MB (conv/BN out)
  float* xB    = (float*)w; w += (long)(N0 / 2) * 128 * 4;  // 16 MB (pooled x)
  float* xroot = (float*)w; w += (long)N0 * 128 * 4;        // 32 MB
  float* agg   = (float*)w; w += (long)N0 * 64 * 4;         // 16 MB (max M*din)
  int*   ei    = (int*)w;   w += (long)2 * Ecnt * 4;        // 8 MB
  float* emask = (float*)w; w += (long)Ecnt * 4;            // 4 MB
  float* score = (float*)w; w += (long)N0 * 4;
  int*   perm  = (int*)w;   w += (long)N0 * 4;
  int*   inv   = (int*)w;   w += (long)N0 * 4;
  float* mean  = (float*)w; w += 128 * 4;
  float* var   = (float*)w; w += 128 * 4;
  float* pn    = (float*)w; w += 64;
  float* rsum  = (float*)w; w += 128 * 256 * 4;
  float* h1    = (float*)w; w += 128 * 128 * 4;
  float* h2    = (float*)w; w += 128 * 64 * 4;

  // init: copy edges (don't mutate inputs), emask = 1, rsum = 0
  copy_i32<<<gdiv(2L * Ecnt, 256), 256, 0, stream>>>(ei, eidx, 2 * Ecnt);
  fill_f32<<<gdiv(Ecnt, 256), 256, 0, stream>>>(emask, 1.0f, Ecnt);
  fill_f32<<<gdiv(128L * 256, 256), 256, 0, stream>>>(rsum, 0.0f, 128 * 256);

  int Mcur = N0, ncur = 512;
  for (int i = 0; i < 5; ++i) {
    const int din = (i == 0) ? 64 : 128;
    const int k = ncur / 2;
    const int Mnew = Mcur / 2;
    const float* xin = (i == 0) ? x_in : xB;

    // agg = segment_sum(x[src] * emask, dst)
    fill_f32<<<gdiv((long)Mcur * din, 256), 256, 0, stream>>>(agg, 0.0f, Mcur * din);
    scatter_edges<<<gdiv((long)Ecnt * (din >> 2), 256), 256, 0, stream>>>(
        xin, ei, emask, agg, Ecnt, din);

    // xroot = x @ Wroot^T ; xA = relu(agg @ Wrel^T + b + xroot)
    gemm_wmma<8><<<Mcur / 128, 256, 0, stream>>>(
        xin, Wroot[i], nullptr, nullptr, xroot, Mcur, din, 128, 0);
    gemm_wmma<8><<<Mcur / 128, 256, 0, stream>>>(
        agg, Wrel[i], cb[i], xroot, xA, Mcur, din, 128, 1);

    // batch-norm (training stats), in place
    bn_stats<<<128, 256, 0, stream>>>(xA, mean, var, Mcur);
    bn_apply<<<gdiv((long)Mcur * 128, 256), 256, 0, stream>>>(xA, mean, var, bng[i], bnb[i], Mcur);

    // top-k pooling
    pnorm_k<<<1, 32, 0, stream>>>(pp[i], pn);
    score_k<<<gdiv(Mcur, 256), 256, 0, stream>>>(xA, pp[i], pn, score, Mcur);
    topk_k<<<B, 256, 0, stream>>>(score, perm, ncur, k);
    gather_gate<<<gdiv((long)Mnew * 128, 256), 256, 0, stream>>>(xA, score, perm, xB, Mnew);

    // relabel + mask edges
    fill_i32<<<gdiv(Mcur, 256), 256, 0, stream>>>(inv, -1, Mcur);
    inv_scatter<<<gdiv(Mnew, 256), 256, 0, stream>>>(inv, perm, Mnew);
    relabel<<<gdiv(Ecnt, 256), 256, 0, stream>>>(ei, emask, inv, Ecnt);

    // readout accumulation (max ++ mean)
    readout_k<<<B, 128, 0, stream>>>(xB, rsum, k);

    Mcur = Mnew; ncur = k;
  }

  // MLP head: lin1 (WMMA), lin2 (WMMA), lin3 + log_softmax
  gemm_wmma<8><<<1, 256, 0, stream>>>(rsum, lin1W, lin1b, nullptr, h1, 128, 256, 128, 1);
  gemm_wmma<4><<<1, 256, 0, stream>>>(h1, lin2W, lin2b, nullptr, h2, 128, 128, 64, 1);
  head_k<<<B, 32, 0, stream>>>(h2, lin3W, lin3b, (float*)d_out);
}
